// PNA_88605175316622
// MI455X (gfx1250) — matrix operand order repaired and checked by reading, not compiled, not run on hardware
//
#include <hip/hip_runtime.h>
#include <hip/hip_bf16.h>

#define NNODES 50000
#define NEDGES 400000
#define DIM 64
#define NLAYERS 5
#define ATOM_C 9
#define ATOM_V 64
#define BOND_C 3
#define BOND_V 8
#define EPSV 1e-5f
#define AVG_LOG 2.19722457733621938f

typedef __bf16 bf16;
typedef __bf16 v16bf __attribute__((ext_vector_type(16)));
typedef __bf16 v8bf  __attribute__((ext_vector_type(8)));
typedef __bf16 v2bf  __attribute__((ext_vector_type(2)));
typedef float  v8f   __attribute__((ext_vector_type(8)));

static __device__ inline v8f wmma_bf16(v16bf a, v16bf b, v8f c) {
  // D = A(16x32 bf16) x B(32x16 bf16) + C(16x16 f32)
  return __builtin_amdgcn_wmma_f32_16x16x32_bf16(false, a, false, b, (short)0, c, false, false);
}
static __device__ inline v16bf cat8(v8bf lo, v8bf hi) {
  return __builtin_shufflevector(lo, hi, 0,1,2,3,4,5,6,7,8,9,10,11,12,13,14,15);
}

// CDNA5 async global->LDS copy of one 16B chunk (tracked by ASYNCcnt).
static __device__ inline void async_copy16(unsigned lds_off, const void* gptr) {
  asm volatile("global_load_async_to_lds_b128 %0, %1, off"
               :: "v"(lds_off), "v"((unsigned long long)(uintptr_t)gptr)
               : "memory");
}
static __device__ inline void wait_async0() {
  asm volatile("s_wait_asynccnt 0" ::: "memory");
}

// ---------------- encoders ----------------
__global__ void k_atom_embed(const int* __restrict__ x, const float* __restrict__ emb,
                             float* __restrict__ h0f, bf16* __restrict__ h0b) {
  int tid = blockIdx.x * blockDim.x + threadIdx.x;
  if (tid >= NNODES * DIM) return;
  int n = tid >> 6, d = tid & 63;
  float s = 0.f;
#pragma unroll
  for (int c = 0; c < ATOM_C; ++c) {
    int v = x[n * ATOM_C + c];
    s += emb[(c * ATOM_V + v) * DIM + d];
  }
  h0f[tid] = s;
  h0b[tid] = (bf16)s;
}

__global__ void k_bond_embed(const int* __restrict__ ea, const float* __restrict__ emb,
                             bf16* __restrict__ eb) {
  int tid = blockIdx.x * blockDim.x + threadIdx.x;
  if (tid >= NEDGES * DIM) return;
  int e = tid >> 6, d = tid & 63;
  float s = 0.f;
#pragma unroll
  for (int c = 0; c < BOND_C; ++c) {
    int v = ea[e * BOND_C + c];
    s += emb[(c * BOND_V + v) * DIM + d];
  }
  eb[tid] = (bf16)s;
}

// ---------------- weight conversion (fp32 row-major -> bf16 n-major/k-contig) ----
__global__ void k_conv_pre(const float* __restrict__ W, bf16* __restrict__ Wt) {
  int idx = blockIdx.x * blockDim.x + threadIdx.x;
  if (idx >= NLAYERS * 3 * DIM * DIM) return;
  int l = idx / (3 * DIM * DIM);
  int rem = idx % (3 * DIM * DIM);
  int blk = rem / (DIM * DIM);
  int nk = rem % (DIM * DIM);
  int nn = nk / DIM;   // output column
  int kk = nk % DIM;   // k index
  Wt[idx] = (bf16)W[(size_t)l * 192 * DIM + (blk * DIM + kk) * DIM + nn];
}

#define POST_PER (3 * 64 * 256 + 64 * 64) // 53248 elements per layer
__global__ void k_conv_post(const float* __restrict__ W, bf16* __restrict__ Wt) {
  int idx = blockIdx.x * blockDim.x + threadIdx.x;
  if (idx >= NLAYERS * POST_PER) return;
  int l = idx / POST_PER, off = idx % POST_PER;
  float v;
  if (off < 3 * 64 * 256) {
    int blk = off / (64 * 256);
    int nk = off % (64 * 256);
    int nn = nk / 256, kk = nk % 256;
    v = W[(size_t)l * 832 * DIM + (blk * 256 + kk) * DIM + nn];
  } else {
    int o2 = off - 3 * 64 * 256;
    int nn = o2 / 64, kk = o2 % 64;
    v = W[(size_t)l * 832 * DIM + (768 + kk) * DIM + nn];
  }
  Wt[idx] = (bf16)v;
}

// ---------------- degree / CSR construction ----------------
__global__ void k_deg_count(const int* __restrict__ dst, int* __restrict__ deg) {
  int e = blockIdx.x * blockDim.x + threadIdx.x;
  if (e >= NEDGES) return;
  atomicAdd(&deg[dst[e]], 1);
}

__global__ void k_scan(const int* __restrict__ deg, int* __restrict__ row_start) {
  __shared__ int buf[1024];
  __shared__ int carry_s;
  int t = threadIdx.x;
  if (t == 0) carry_s = 0;
  __syncthreads();
  for (int base = 0; base < NNODES; base += 1024) {
    int i = base + t;
    int v = (i < NNODES) ? deg[i] : 0;
    buf[t] = v;
    __syncthreads();
    for (int off = 1; off < 1024; off <<= 1) {
      int xv = (t >= off) ? buf[t - off] : 0;
      __syncthreads();
      buf[t] += xv;
      __syncthreads();
    }
    int carry = carry_s;
    if (i < NNODES) row_start[i] = carry + buf[t] - v;  // exclusive
    __syncthreads();
    if (t == 1023) carry_s = carry + buf[1023];
    __syncthreads();
  }
  if (t == 0) row_start[NNODES] = carry_s;
}

__global__ void k_deg_derive(const int* __restrict__ deg, float* __restrict__ degc,
                             float* __restrict__ amp, float* __restrict__ att) {
  int n = blockIdx.x * blockDim.x + threadIdx.x;
  if (n >= NNODES) return;
  float d = (float)deg[n];
  float dc = fmaxf(d, 1.f);
  float ld = logf(dc + 1.f);
  degc[n] = dc;
  amp[n] = ld / AVG_LOG;
  att[n] = AVG_LOG / ld;
}

__global__ void k_csr_fill(const int* __restrict__ src, const int* __restrict__ dst,
                           const int* __restrict__ row_start, int* __restrict__ cursor,
                           int* __restrict__ csr_eid, int* __restrict__ csr_src) {
  int e = blockIdx.x * blockDim.x + threadIdx.x;
  if (e >= NEDGES) return;
  int d = dst[e];
  int pos = atomicAdd(&cursor[d], 1);
  int idx = row_start[d] + pos;
  csr_eid[idx] = e;
  csr_src[idx] = src[e];
}

// ---------------- WMMA GEMM: C[M,64] = A[M,64](bf16) @ W[64,64] ----------------
// Wt transposed bf16 (Wt[n*64+k]); staged to LDS with async copies, one wave per
// 16x64 output tile, 4 waves / block.
template <bool OUT_BF16, bool HAS_BIAS>
__global__ void k_gemm64(const bf16* __restrict__ A, const bf16* __restrict__ Wt,
                         const float* __restrict__ bias, void* __restrict__ Cout, int M) {
  extern __shared__ __align__(32) char smem_raw[];
  bf16* sW = (bf16*)smem_raw;                 // 64*64 bf16 = 8KB
  {
    unsigned base = (unsigned)(uintptr_t)sW;
    const char* g = (const char*)Wt;
#pragma unroll
    for (int k = 0; k < 4; ++k) {             // 512 chunks of 16B / 128 threads
      int c = (int)threadIdx.x + k * 128;
      async_copy16(base + (unsigned)c * 16u, g + (size_t)c * 16);
    }
    wait_async0();
  }
  __syncthreads();

  int wid = threadIdx.x >> 5;
  int lane = threadIdx.x & 31;
  int m0 = (blockIdx.x * 4 + wid) * 16;
  if (m0 >= M) return;
  int r = lane & 15;
  int kh = lane >> 4;
  // A tile (16-bit A 16x32): lane<16 row=lane K0..15; lane>=16 K16..31 (contig)
  const bf16* arow = A + (size_t)(m0 + r) * DIM + kh * 16;
  v16bf a0 = cat8(*(const v8bf*)(arow + 0),  *(const v8bf*)(arow + 8));
  v16bf a1 = cat8(*(const v8bf*)(arow + 32), *(const v8bf*)(arow + 40));
  // B tile (16-bit B 32x16): lane<16 col=lane K0..15; lane>=16 K16..31 (contig)
  int bn = r, bk = kh * 16;
  v8f acc[4];
#pragma unroll
  for (int t = 0; t < 4; ++t) {
    const bf16* wrow = sW + (t * 16 + bn) * DIM + bk;
    v16bf b0 = *(const v16bf*)(wrow);
    v16bf b1 = *(const v16bf*)(wrow + 32);
    v8f c = {0.f, 0.f, 0.f, 0.f, 0.f, 0.f, 0.f, 0.f};
    c = wmma_bf16(a0, b0, c);
    c = wmma_bf16(a1, b1, c);
    acc[t] = c;
  }
  int mbase = m0 + kh * 8;
#pragma unroll
  for (int t = 0; t < 4; ++t) {
    int col = t * 16 + bn;
    float bv = HAS_BIAS ? bias[col] : 0.f;
#pragma unroll
    for (int j = 0; j < 8; ++j) {
      float val = acc[t][j] + bv;
      size_t o = (size_t)(mbase + j) * DIM + col;
      if (OUT_BF16) ((bf16*)Cout)[o] = (bf16)val;
      else          ((float*)Cout)[o] = val;
    }
  }
}

// ---------------- aggregation: one wave per node, 2 features per lane --------
__global__ void k_agg(const float* __restrict__ P, const bf16* __restrict__ Qb,
                      const bf16* __restrict__ Rb, const int* __restrict__ row_start,
                      const int* __restrict__ csr_eid, const int* __restrict__ csr_src,
                      const float* __restrict__ degc, bf16* __restrict__ aggb) {
  int n = blockIdx.x * (blockDim.x >> 5) + (threadIdx.x >> 5);
  int lane = threadIdx.x & 31;
  if (n >= NNODES) return;
  int f = lane * 2;
  float p0 = P[(size_t)n * DIM + f], p1 = P[(size_t)n * DIM + f + 1]; // bias folded in
  int beg = row_start[n], end = row_start[n + 1];
  float s0 = 0.f, s1 = 0.f, ss0 = 0.f, ss1 = 0.f;
  float mx0 = -3.4e38f, mx1 = -3.4e38f, mn0 = 3.4e38f, mn1 = 3.4e38f;
  for (int i = beg; i < end; ++i) {
    int eid = csr_eid[i], sv = csr_src[i];
    v2bf qq = *(const v2bf*)(Qb + (size_t)sv * DIM + f);
    v2bf rr = *(const v2bf*)(Rb + (size_t)eid * DIM + f);
    float m0v = p0 + (float)qq[0] + (float)rr[0];
    float m1v = p1 + (float)qq[1] + (float)rr[1];
    s0 += m0v; s1 += m1v;
    ss0 += m0v * m0v; ss1 += m1v * m1v;
    mx0 = fmaxf(mx0, m0v); mx1 = fmaxf(mx1, m1v);
    mn0 = fminf(mn0, m0v); mn1 = fminf(mn1, m1v);
  }
  float dc = degc[n];
  float mean0 = s0 / dc, mean1 = s1 / dc;
  float var0 = fmaxf(ss0 / dc - mean0 * mean0, 0.f);
  float var1 = fmaxf(ss1 / dc - mean1 * mean1, 0.f);
  float sd0 = sqrtf(var0 + EPSV), sd1 = sqrtf(var1 + EPSV);
  if (beg == end) { mx0 = mx1 = mn0 = mn1 = 0.f; }
  bf16* o = aggb + (size_t)n * 256;
  o[f]       = (bf16)mean0; o[f + 1]       = (bf16)mean1;
  o[64 + f]  = (bf16)mx0;   o[64 + f + 1]  = (bf16)mx1;
  o[128 + f] = (bf16)mn0;   o[128 + f + 1] = (bf16)mn1;
  o[192 + f] = (bf16)sd0;   o[192 + f + 1] = (bf16)sd1;
}

// ---------------- posttrans + scalers + residual (fused WMMA) ----------------
// out = agg@Wa + amp*(agg@Wb) + att*(agg@Wc) + h0@Wd + b + hprev
// 16 waves / block; per-layer weights (104KB bf16) staged to LDS via async copy;
// all A fragments preloaded so the WMMA loop has only LDS B reads.
__global__ void k_post(const bf16* __restrict__ aggb, const bf16* __restrict__ h0b,
                       const bf16* __restrict__ Wp, const float* __restrict__ postb,
                       const float* __restrict__ amp, const float* __restrict__ att,
                       const float* __restrict__ hprev, float* __restrict__ hout) {
  extern __shared__ __align__(32) char smem_raw[];
  bf16* sW = (bf16*)smem_raw;                 // POST_PER bf16 = 106496 B
  {
    unsigned base = (unsigned)(uintptr_t)sW;
    const char* g = (const char*)Wp;
#pragma unroll
    for (int k = 0; k < 13; ++k) {            // 6656 chunks of 16B / 512 threads
      int c = (int)threadIdx.x + k * 512;
      async_copy16(base + (unsigned)c * 16u, g + (size_t)c * 16);
    }
    wait_async0();
  }
  __syncthreads();

  int wid = threadIdx.x >> 5;
  int lane = threadIdx.x & 31;
  int m0 = (blockIdx.x * 16 + wid) * 16;
  if (m0 >= NNODES) return;
  int r = lane & 15, kh = lane >> 4;
  int bn = r, bk = kh * 16;
  const bf16* sWa = sW;
  const bf16* sWb = sW + 64 * 256;
  const bf16* sWc = sW + 2 * 64 * 256;
  const bf16* sWd = sW + 3 * 64 * 256;

  // Preload all A fragments (agg: 8 x K32, h0: 2 x K32).
  const bf16* arow = aggb + (size_t)(m0 + r) * 256 + kh * 16;
  v16bf aF[8];
#pragma unroll
  for (int kt = 0; kt < 8; ++kt)
    aF[kt] = cat8(*(const v8bf*)(arow + kt * 32), *(const v8bf*)(arow + kt * 32 + 8));
  const bf16* hrow = h0b + (size_t)(m0 + r) * 64 + kh * 16;
  v16bf hF0 = cat8(*(const v8bf*)(hrow + 0),  *(const v8bf*)(hrow + 8));
  v16bf hF1 = cat8(*(const v8bf*)(hrow + 32), *(const v8bf*)(hrow + 40));

  v8f zero = {0.f, 0.f, 0.f, 0.f, 0.f, 0.f, 0.f, 0.f};
  v8f aA[4] = {zero, zero, zero, zero};
  v8f aB[4] = {zero, zero, zero, zero};
  v8f aC[4] = {zero, zero, zero, zero};

  // h0 @ Wd first (frees hF0/hF1 early)
#pragma unroll
  for (int t = 0; t < 4; ++t) {
    int wo = (t * 16 + bn) * 64 + bk;
    aA[t] = wmma_bf16(hF0, *(const v16bf*)(sWd + wo), aA[t]);
    aA[t] = wmma_bf16(hF1, *(const v16bf*)(sWd + wo + 32), aA[t]);
  }
  // agg @ {Wa, Wb, Wc}
#pragma unroll
  for (int kt = 0; kt < 8; ++kt) {
#pragma unroll
    for (int t = 0; t < 4; ++t) {
      int wo = (t * 16 + bn) * 256 + kt * 32 + bk;
      aA[t] = wmma_bf16(aF[kt], *(const v16bf*)(sWa + wo), aA[t]);
      aB[t] = wmma_bf16(aF[kt], *(const v16bf*)(sWb + wo), aB[t]);
      aC[t] = wmma_bf16(aF[kt], *(const v16bf*)(sWc + wo), aC[t]);
    }
  }

  int mbase = m0 + kh * 8;
#pragma unroll
  for (int t = 0; t < 4; ++t) {
    int col = t * 16 + bn;
    float bv = postb[col];
#pragma unroll
    for (int j = 0; j < 8; ++j) {
      int m = mbase + j;
      float val = aA[t][j] + amp[m] * aB[t][j] + att[m] * aC[t][j] + bv
                + hprev[(size_t)m * 64 + col];
      hout[(size_t)m * 64 + col] = val;
    }
  }
}

// ---------------- launch ----------------
extern "C" void kernel_launch(void* const* d_in, const int* in_sizes, int n_in,
                              void* d_out, int out_size, void* d_ws, size_t ws_size,
                              hipStream_t stream) {
  (void)in_sizes; (void)n_in; (void)out_size; (void)ws_size;
  const int*   x        = (const int*)d_in[0];
  const int*   ei       = (const int*)d_in[1];
  const int*   ea       = (const int*)d_in[2];
  const float* atom_emb = (const float*)d_in[3];
  const float* bond_emb = (const float*)d_in[4];
  const float* pre_W    = (const float*)d_in[5];
  const float* pre_b    = (const float*)d_in[6];
  const float* post_W   = (const float*)d_in[7];
  const float* post_b   = (const float*)d_in[8];
  const int* srcI = ei;
  const int* dstI = ei + NEDGES;

  char* p = (char*)d_ws;
  auto alloc = [&](size_t bytes) {
    char* r = p;
    p += (bytes + 255) & ~(size_t)255;
    return r;
  };
  float* h0f   = (float*)alloc((size_t)NNODES * DIM * 4);
  bf16*  h0b   = (bf16*) alloc((size_t)NNODES * DIM * 2);
  bf16*  eb    = (bf16*) alloc((size_t)NEDGES * DIM * 2);
  float* Pf    = (float*)alloc((size_t)NNODES * DIM * 4);
  bf16*  Qb    = (bf16*) alloc((size_t)NNODES * DIM * 2);
  bf16*  Rb    = (bf16*) alloc((size_t)NEDGES * DIM * 2);
  bf16*  aggb  = (bf16*) alloc((size_t)NNODES * 256 * 2);
  float* hbuf  = (float*)alloc((size_t)NNODES * DIM * 4);
  int*   deg   = (int*)  alloc((size_t)NNODES * 4);
  int*   rowst = (int*)  alloc((size_t)(NNODES + 1) * 4);
  int*   curs  = (int*)  alloc((size_t)NNODES * 4);
  float* degc  = (float*)alloc((size_t)NNODES * 4);
  float* ampv  = (float*)alloc((size_t)NNODES * 4);
  float* attv  = (float*)alloc((size_t)NNODES * 4);
  int*   ceid  = (int*)  alloc((size_t)NEDGES * 4);
  int*   csrc  = (int*)  alloc((size_t)NEDGES * 4);
  bf16*  preWt = (bf16*) alloc((size_t)NLAYERS * 3 * DIM * DIM * 2);
  bf16*  postWt= (bf16*) alloc((size_t)NLAYERS * POST_PER * 2);

  hipMemsetAsync(deg, 0, (size_t)NNODES * 4, stream);
  hipMemsetAsync(curs, 0, (size_t)NNODES * 4, stream);

  k_atom_embed<<<(NNODES * DIM + 255) / 256, 256, 0, stream>>>(x, atom_emb, h0f, h0b);
  k_bond_embed<<<(NEDGES * DIM + 255) / 256, 256, 0, stream>>>(ea, bond_emb, eb);
  k_conv_pre<<<(NLAYERS * 3 * DIM * DIM + 255) / 256, 256, 0, stream>>>(pre_W, preWt);
  k_conv_post<<<(NLAYERS * POST_PER + 255) / 256, 256, 0, stream>>>(post_W, postWt);
  k_deg_count<<<(NEDGES + 255) / 256, 256, 0, stream>>>(dstI, deg);
  k_scan<<<1, 1024, 0, stream>>>(deg, rowst);
  k_deg_derive<<<(NNODES + 255) / 256, 256, 0, stream>>>(deg, degc, ampv, attv);
  k_csr_fill<<<(NEDGES + 255) / 256, 256, 0, stream>>>(srcI, dstI, rowst, curs, ceid, csrc);

  const int gN = ((NNODES + 15) / 16 + 3) / 4;   // 782 blocks of 4 waves
  const int gE = ((NEDGES + 15) / 16 + 3) / 4;   // 6250
  const int gA = (NNODES + 7) / 8;               // 6250 (wave per node)
  const int gP = ((NNODES + 15) / 16 + 15) / 16; // 196 blocks of 16 waves

  const size_t lds64 = (size_t)DIM * DIM * 2;        // 8KB
  const size_t ldsP  = (size_t)POST_PER * 2;         // 106496B

  for (int l = 0; l < NLAYERS; ++l) {
    const bf16* Wpre = preWt + (size_t)l * 3 * DIM * DIM;
    k_gemm64<false, true ><<<gN, 128, lds64, stream>>>(h0b, Wpre + 0,             pre_b + l * DIM, Pf, NNODES);
    k_gemm64<true,  false><<<gN, 128, lds64, stream>>>(h0b, Wpre + DIM * DIM,     nullptr,         Qb, NNODES);
    k_gemm64<true,  false><<<gE, 128, lds64, stream>>>(eb,  Wpre + 2 * DIM * DIM, nullptr,         Rb, NEDGES);
    k_agg<<<gA, 256, 0, stream>>>(Pf, Qb, Rb, rowst, ceid, csrc, degc, aggb);
    const float* hp = (l == 0) ? h0f : hbuf;
    float* ho = (l == NLAYERS - 1) ? (float*)d_out : hbuf;
    k_post<<<gP, 512, ldsP, stream>>>(aggb, h0b, postWt + (size_t)l * POST_PER,
                                      post_b + l * DIM, ampv, attv, hp, ho);
  }
}
